// MoELayer_44890998178064
// MI455X (gfx1250) — matrix operand order repaired and checked by reading, not compiled, not run on hardware
//
#include <hip/hip_runtime.h>
#include <hip/hip_bf16.h>

// MoE layer for MI455X (gfx1250, wave32, WMMA).
// Workspace use (~369 MB, assumed available in d_ws):
//   probs, topk, ranks, hists, disp(bf16), W1t/W2t(bf16), H(bf16), Y(f32)

namespace {

constexpr int kB = 4, kT = 2048, kC = 1024, kF = 4096, kE = 8, kK = 2;
constexpr int kN   = kB * kT;     // 8192 tokens
constexpr int kS   = kN * kK;     // 16384 dispatch slots
constexpr int kCap = 2048;        // capacity per expert
constexpr int kPosBlocks = kS / 256;  // 64

typedef __attribute__((ext_vector_type(16))) __bf16 bf16x16;
typedef __attribute__((ext_vector_type(8)))  float  floatx8;

union FragU { bf16x16 v; uint4 q[2]; };

__device__ __forceinline__ float gelu_tanh(float x) {
  const float kA = 0.7978845608028654f;  // sqrt(2/pi)
  float t = kA * (x + 0.044715f * x * x * x);
  return 0.5f * x * (1.0f + tanhf(t));
}

// ---------------------------------------------------------------- gating ----
__global__ __launch_bounds__(256) void gating_kernel(
    const float* __restrict__ x, const float* __restrict__ wg,
    const float* __restrict__ bg, float* __restrict__ probs,
    int* __restrict__ topk_idx, float* __restrict__ topk_w) {
  const int tok = blockIdx.x;
  __shared__ float sX[kC];
  const float* xrow = x + (size_t)tok * kC;
  for (int i = threadIdx.x; i < kC; i += 256) sX[i] = xrow[i];
  __syncthreads();
  const int wave = threadIdx.x >> 5, lane = threadIdx.x & 31;
  // wave == expert id (8 waves, 8 experts)
  float acc = 0.f;
  for (int c = lane; c < kC; c += 32) acc += sX[c] * wg[c * kE + wave];
  for (int off = 16; off; off >>= 1) acc += __shfl_xor(acc, off);
  __shared__ float sL[kE];
  if (lane == 0) sL[wave] = acc + bg[wave];
  __syncthreads();
  if (threadIdx.x == 0) {
    float p[kE], mx = -3.0e38f;
    for (int e = 0; e < kE; ++e) { p[e] = sL[e]; mx = fmaxf(mx, p[e]); }
    float s = 0.f;
    for (int e = 0; e < kE; ++e) { p[e] = expf(p[e] - mx); s += p[e]; }
    float inv = 1.f / s;
    for (int e = 0; e < kE; ++e) { p[e] *= inv; probs[(size_t)tok * kE + e] = p[e]; }
    int i0 = 0;
    for (int e = 1; e < kE; ++e) if (p[e] > p[i0]) i0 = e;      // first max on ties
    int i1 = (i0 == 0) ? 1 : 0;
    for (int e = 0; e < kE; ++e) if (e != i1 && e != i0 && p[e] > p[i1]) i1 = e;
    float w0 = p[i0], w1 = p[i1], ws = w0 + w1 + 1e-9f;
    topk_idx[tok * 2 + 0] = i0; topk_idx[tok * 2 + 1] = i1;
    topk_w[tok * 2 + 0] = w0 / ws; topk_w[tok * 2 + 1] = w1 / ws;
  }
}

// --------------------------------------- weight transpose + f32->bf16 ------
// in: [E][R][Cc] f32 -> out: [E][Cc][R] bf16 (K-contiguous rows for GEMM B^T)
__global__ __launch_bounds__(256) void transconv_kernel(
    const float* __restrict__ in, __bf16* __restrict__ out, int R, int Cc) {
  const int e = blockIdx.z;
  const int c0 = blockIdx.x * 32, r0 = blockIdx.y * 32;
  __shared__ float tile[32][33];
  const float* ip = in + (size_t)e * R * Cc;
  for (int rr = threadIdx.y; rr < 32; rr += 8)
    tile[rr][threadIdx.x] = ip[(size_t)(r0 + rr) * Cc + c0 + threadIdx.x];
  __syncthreads();
  __bf16* op = out + (size_t)e * R * Cc;
  for (int cc = threadIdx.y; cc < 32; cc += 8)
    op[(size_t)(c0 + cc) * R + r0 + threadIdx.x] = (__bf16)tile[threadIdx.x][cc];
}

// ------------------------------------------------ FCFS capacity routing ----
__global__ __launch_bounds__(256) void pos1_kernel(
    const int* __restrict__ eflat, int* __restrict__ rank_local,
    int* __restrict__ block_hist) {
  const int s = blockIdx.x * 256 + threadIdx.x;
  const int e = eflat[s];
  const int lane = threadIdx.x & 31, wave = threadIdx.x >> 5;
  unsigned same = 0u;
  for (int ee = 0; ee < kE; ++ee) {
    unsigned b = (unsigned)__ballot(e == ee);
    if (e == ee) same = b;
  }
  const int rankw = __popc(same & ((1u << lane) - 1u));  // rank within wave
  const int cnt   = __popc(same);                        // wave count for e
  __shared__ int base[kE];
  if (threadIdx.x < kE) base[threadIdx.x] = 0;
  __syncthreads();
  int myrank = 0;
  for (int w = 0; w < 8; ++w) {        // waves commit in slot order
    if (wave == w) {
      myrank = base[e] + rankw;
      if (rankw == 0) base[e] += cnt;  // one lane per distinct e
    }
    __syncthreads();
  }
  rank_local[s] = myrank;
  if (threadIdx.x < kE) block_hist[blockIdx.x * kE + threadIdx.x] = base[threadIdx.x];
}

__global__ void pos2_kernel(const int* __restrict__ block_hist,
                            int* __restrict__ block_off,
                            float* __restrict__ expert_load) {
  const int e = threadIdx.x;
  if (e < kE) {
    int run = 0;
    for (int b = 0; b < kPosBlocks; ++b) {
      block_off[b * kE + e] = run;
      run += block_hist[b * kE + e];
    }
    expert_load[e] = (float)(run < kCap ? run : kCap);
  }
}

// ----------------------------------------------------------- dispatch ------
__global__ __launch_bounds__(256) void dispatch_kernel(
    const float* __restrict__ x, const int* __restrict__ eflat,
    const int* __restrict__ rank_local, const int* __restrict__ block_off,
    int* __restrict__ pos_final, __bf16* __restrict__ disp) {
  const int s = blockIdx.x;
  const int e = eflat[s];
  const int pos = block_off[(s >> 8) * kE + e] + rank_local[s];
  const bool keep = pos < kCap;
  if (threadIdx.x == 0) pos_final[s] = keep ? pos : -1;
  if (!keep) return;
  const int tok = s >> 1;  // kK == 2
  const float* src = x + (size_t)tok * kC;
  __bf16* dst = disp + ((size_t)e * kCap + pos) * kC;
  const int i = threadIdx.x * 4;
  float4 v = *(const float4*)(src + i);
  union { __bf16 h[4]; unsigned long long u; } pk;
  pk.h[0] = (__bf16)v.x; pk.h[1] = (__bf16)v.y;
  pk.h[2] = (__bf16)v.z; pk.h[3] = (__bf16)v.w;
  *(unsigned long long*)(dst + i) = pk.u;
}

// ------------------------------------------------- bf16 WMMA grouped GEMM --
// A: [E][M][Kd] bf16 row-major, Bt: [E][N][Kd] bf16 (K-contiguous),
// Out = A * Bt^T + bias;  MODE 0: gelu -> bf16 (H), MODE 1: f32 (Y).
template <int MODE>
__global__ __launch_bounds__(256) void gemm_kernel(
    const __bf16* __restrict__ A, const __bf16* __restrict__ Bt,
    const float* __restrict__ bias, void* __restrict__ Out,
    int M, int N, int Kd) {
  const int e = blockIdx.z;
  const __bf16* Ae = A  + (size_t)e * M * Kd;
  const __bf16* Be = Bt + (size_t)e * N * Kd;
  const float*  bv = bias + (size_t)e * N;

  __shared__ __attribute__((aligned(16))) __bf16 sA[128][40];
  __shared__ __attribute__((aligned(16))) __bf16 sB[128][40];

  const int m0 = blockIdx.y * 128, n0 = blockIdx.x * 128;
  const int t = threadIdx.x;
  const int lr = t >> 1, lh = t & 1;                    // tile row / 16-col half
  const size_t aRow = (size_t)(m0 + lr) * Kd;
  const size_t bRow = (size_t)(n0 + lr) * Kd;

  const int lane = t & 31, wave = t >> 5;
  const int wm = (wave >> 2) * 64;                      // 2 waves along M
  const int wn = (wave & 3) * 32;                       // 4 waves along N
  const int half = lane >> 4, r = lane & 15;

  floatx8 acc[4][2] = {};

  for (int k0 = 0; k0 < Kd; k0 += 32) {
    const uint4* pa = (const uint4*)(Ae + aRow + k0 + lh * 16);
    const uint4* pb = (const uint4*)(Be + bRow + k0 + lh * 16);
    uint4 qa0 = pa[0], qa1 = pa[1];
    uint4 qb0 = pb[0], qb1 = pb[1];
    if (k0 + 32 < Kd) {                                 // -> global_prefetch_b8
      __builtin_prefetch((const void*)(Ae + aRow + k0 + 32 + lh * 16), 0, 1);
      __builtin_prefetch((const void*)(Be + bRow + k0 + 32 + lh * 16), 0, 1);
    }
    __syncthreads();
    *(uint4*)&sA[lr][lh * 16 + 0] = qa0;
    *(uint4*)&sA[lr][lh * 16 + 8] = qa1;
    *(uint4*)&sB[lr][lh * 16 + 0] = qb0;
    *(uint4*)&sB[lr][lh * 16 + 8] = qb1;
    __syncthreads();

    // A fragment (16x32): lane<16 holds K {0..7,16..23}, lane>=16 {8..15,24..31}
    FragU fa[4], fb[2];
#pragma unroll
    for (int i = 0; i < 4; ++i) {
      const __bf16* p = &sA[wm + i * 16 + r][half * 8];
      fa[i].q[0] = *(const uint4*)p;
      fa[i].q[1] = *(const uint4*)(p + 16);
    }
    // B fragment (32x16): lane<16 holds K {0..7,16..23}, lane>=16 {8..15,24..31}
#pragma unroll
    for (int j = 0; j < 2; ++j) {
      const __bf16* p = &sB[wn + j * 16 + r][half * 8];
      fb[j].q[0] = *(const uint4*)p;
      fb[j].q[1] = *(const uint4*)(p + 16);
    }
#pragma unroll
    for (int i = 0; i < 4; ++i)
#pragma unroll
      for (int j = 0; j < 2; ++j)
        acc[i][j] = __builtin_amdgcn_wmma_f32_16x16x32_bf16(
            false, fa[i].v, false, fb[j].v, (short)0, acc[i][j], false, false);
  }

  // C/D layout: VGPR rr -> row rr + 8*half, col = lane&15
#pragma unroll
  for (int j = 0; j < 2; ++j) {
    const int col = n0 + wn + j * 16 + (lane & 15);
    const float bb = bv[col];
#pragma unroll
    for (int i = 0; i < 4; ++i) {
      const int rowBase = m0 + wm + i * 16 + 8 * half;
#pragma unroll
      for (int rr = 0; rr < 8; ++rr) {
        const float v = acc[i][j][rr] + bb;
        const size_t idx = (size_t)e * M * N + (size_t)(rowBase + rr) * N + col;
        if (MODE == 0) ((__bf16*)Out)[idx] = (__bf16)gelu_tanh(v);
        else           ((float*)Out)[idx]  = v;
      }
    }
  }
}

// ------------------------------------------------------------ combine ------
__global__ __launch_bounds__(256) void combine_kernel(
    const float* __restrict__ Y, const int* __restrict__ eflat,
    const int* __restrict__ pos_final, const float* __restrict__ topk_w,
    float* __restrict__ out) {
  const int tok = blockIdx.x;
  const int i = threadIdx.x * 4;
  float4 a = make_float4(0.f, 0.f, 0.f, 0.f);
  for (int k = 0; k < kK; ++k) {       // fixed order -> deterministic
    const int s = tok * kK + k;
    const int pos = pos_final[s];
    if (pos < 0) continue;             // block-uniform branch
    const int e = eflat[s];
    const float w = topk_w[s];
    float4 v = *(const float4*)(Y + ((size_t)e * kCap + pos) * kC + i);
    a.x += w * v.x; a.y += w * v.y; a.z += w * v.z; a.w += w * v.w;
  }
  *(float4*)(out + (size_t)tok * kC + i) = a;
}

// ------------------------------------------------------------ aux loss -----
__global__ void aux_kernel(const float* __restrict__ probs,
                           const float* __restrict__ expert_load,
                           float* __restrict__ out_aux) {
  const int e = threadIdx.x >> 5, lane = threadIdx.x & 31;
  float s = 0.f;
  for (int i = lane; i < kN; i += 32) s += probs[(size_t)i * kE + e];
  for (int off = 16; off; off >>= 1) s += __shfl_xor(s, off);
  __shared__ float fe[kE];
  if (lane == 0) fe[e] = s / (float)kN;
  __syncthreads();
  if (threadIdx.x == 0) {
    float te = 0.f, tl = 0.f;
    for (int k = 0; k < kE; ++k) { te += fe[k]; tl += expert_load[k]; }
    float acc = 0.f;
    for (int k = 0; k < kE; ++k) {
      float d = fe[k] / (te + 1e-9f) - expert_load[k] / (tl + 1e-9f);
      acc += d * d;
    }
    *out_aux = 0.01f * (acc / (float)kE);
  }
}

}  // namespace

// ---------------------------------------------------------------------------
extern "C" void kernel_launch(void* const* d_in, const int* in_sizes, int n_in,
                              void* d_out, int out_size, void* d_ws, size_t ws_size,
                              hipStream_t stream) {
  const float* x  = (const float*)d_in[0];
  const float* wg = (const float*)d_in[1];
  const float* bg = (const float*)d_in[2];
  const float* W1 = (const float*)d_in[3];
  const float* b1 = (const float*)d_in[4];
  const float* W2 = (const float*)d_in[5];
  const float* b2 = (const float*)d_in[6];
  float* out = (float*)d_out;

  char* w = (char*)d_ws;
  auto take = [&](size_t bytes) -> void* {
    void* p = (void*)w;
    w += (bytes + 255) & ~(size_t)255;
    return p;
  };
  float*  probs       = (float*) take(sizeof(float) * (size_t)kN * kE);
  int*    topk_idx    = (int*)   take(sizeof(int)   * (size_t)kS);
  float*  topk_w      = (float*) take(sizeof(float) * (size_t)kS);
  int*    rank_local  = (int*)   take(sizeof(int)   * (size_t)kS);
  int*    block_hist  = (int*)   take(sizeof(int)   * (size_t)kPosBlocks * kE);
  int*    block_off   = (int*)   take(sizeof(int)   * (size_t)kPosBlocks * kE);
  float*  expert_load = (float*) take(sizeof(float) * kE);
  int*    pos_final   = (int*)   take(sizeof(int)   * (size_t)kS);
  __bf16* disp        = (__bf16*)take(sizeof(__bf16) * (size_t)kE * kCap * kC);
  __bf16* W1t         = (__bf16*)take(sizeof(__bf16) * (size_t)kE * kC * kF);
  __bf16* W2t         = (__bf16*)take(sizeof(__bf16) * (size_t)kE * kF * kC);
  __bf16* H           = (__bf16*)take(sizeof(__bf16) * (size_t)kE * kCap * kF);
  float*  Y           = (float*) take(sizeof(float)  * (size_t)kE * kCap * kC);

  gating_kernel<<<kN, 256, 0, stream>>>(x, wg, bg, probs, topk_idx, topk_w);
  transconv_kernel<<<dim3(kF / 32, kC / 32, kE), dim3(32, 8), 0, stream>>>(W1, W1t, kC, kF);
  transconv_kernel<<<dim3(kC / 32, kF / 32, kE), dim3(32, 8), 0, stream>>>(W2, W2t, kF, kC);
  pos1_kernel<<<kPosBlocks, 256, 0, stream>>>(topk_idx, rank_local, block_hist);
  pos2_kernel<<<1, 32, 0, stream>>>(block_hist, block_off, expert_load);
  dispatch_kernel<<<kS, 256, 0, stream>>>(x, topk_idx, rank_local, block_off, pos_final, disp);
  gemm_kernel<0><<<dim3(kF / 128, kCap / 128, kE), 256, 0, stream>>>(disp, W1t, b1, (void*)H, kCap, kF, kC);
  gemm_kernel<1><<<dim3(kC / 128, kCap / 128, kE), 256, 0, stream>>>(H, W2t, b2, (void*)Y, kCap, kC, kF);
  combine_kernel<<<kN, 256, 0, stream>>>(Y, topk_idx, pos_final, topk_w, out);
  aux_kernel<<<1, 256, 0, stream>>>(probs, expert_load, out + (size_t)kN * kC);

  (void)in_sizes; (void)n_in; (void)out_size; (void)ws_size;
}
